// HW_Block_Series_26774826123459
// MI455X (gfx1250) — matrix operand order repaired and checked by reading, not compiled, not run on hardware
//
#include <hip/hip_runtime.h>
#include <hip/hip_bf16.h>

typedef __attribute__((ext_vector_type(16))) _Float16 v16h;
typedef __attribute__((ext_vector_type(8)))  _Float16 v8h;
typedef __attribute__((ext_vector_type(4)))  _Float16 v4h;
typedef __attribute__((ext_vector_type(8)))  float    v8f;

#define Bc   2
#define Cc   128
#define Hc   96
#define Wc   96
#define HWc  9216
#define Dc   64
#define Kc   4
#define Nst  16
#define Rc   4
#define Lc   1152
#define BNc  16
#define SIMAM_LAM 1e-4f
#define LN_EPS    1e-6f

static __device__ __forceinline__ float dsigmoid(float x) { return 1.f / (1.f + __expf(-x)); }
static __device__ __forceinline__ float dsilu(float x)    { return x * dsigmoid(x); }

// ---------------------------------------------------------------------------
// Batched WMMA GEMM:  C[M,N] = act(A[M,K] * B[K,N] + bias)
// Requirements (enforced by caller): N % 128 == 0, K % 64 == 0.
// M may be ragged (clamp-load + select, store guard).
// Block: 256 thr = 8 waves; tile 32(M) x 128(N); K chunk 64.
// Wave tile: 16(M) x 64(N) -> 4 accumulators, 8 wmma per K-chunk.
// Software-pipelined double-buffered LDS staging:
//   loads(k+1) -> compute(k) -> stores(k+1) -> barrier.
// ---------------------------------------------------------------------------
#define GTM 32
#define GTN 128
#define GTK 64

__global__ __launch_bounds__(256)
void k_wmma_gemm(const float* __restrict__ A, int lda, long sA, int amod,
                 const float* __restrict__ Bm, int ldb, long sB,
                 float* __restrict__ Cm, int ldc, long sC,
                 const float* __restrict__ bias, int act,
                 int M, int N, int Kd)
{
    __shared__ _Float16 Ash[2][GTM][GTK + 8];   // [buf][m][k], 144B rows
    __shared__ _Float16 Bsh[2][GTN][GTK + 8];   // [buf][n][k] (transposed)

    int bb = blockIdx.z;
    const float* Ab = A  + (long)(bb % amod) * sA;
    const float* Bb = Bm + (long)bb * sB;
    float*       Cb = Cm + (long)bb * sC;

    int m0 = blockIdx.y * GTM;
    int n0 = blockIdx.x * GTN;
    int tid  = threadIdx.x;
    int lane = tid & 31;
    int wv   = tid >> 5;
    int wm   = (wv >> 1) * 16;       // wave row offset: 0/16/32/48
    int wn   = (wv & 1)  * 64;       // wave col offset: 0/64
    int hs   = lane >> 4;            // half-select
    int lr   = lane & 15;
    int arow = wm + lr;

#if defined(__has_builtin)
#if __has_builtin(__builtin_amdgcn_s_wait_tensorcnt)
    __builtin_amdgcn_s_wait_tensorcnt(0);
#endif
#endif

    v8f zz = {};
    v8f acc[4] = {zz, zz, zz, zz};

    float4 fa[2];   // staged A values (chunk in flight)
    float4 fb[8];   // staged B values

    // ---- batched global loads for one K-chunk (no waits interleaved) ----
    auto loadAB = [&](int kc) {
        #pragma unroll
        for (int p = 0; p < 2; ++p) {
            int v  = tid + p * 256;          // 0..511
            int mr = v & 31;
            int kk = (v >> 5) * 4;           // 0..60
            int gm = m0 + mr;
            int cm = gm < M ? gm : M - 1;
            fa[p] = *(const float4*)(Ab + (long)cm * lda + kc + kk);
        }
        #pragma unroll
        for (int p = 0; p < 8; ++p) {
            int v  = tid + p * 256;          // 0..2047
            int n4 = (v & 31) * 4;           // 0..124
            int kk = v >> 5;                 // 0..63
            fb[p] = *(const float4*)(Bb + (long)(kc + kk) * ldb + n0 + n4);
        }
    };
    // ---- convert + LDS stores into buffer `buf` ----
    auto storeAB = [&](int buf) {
        #pragma unroll
        for (int p = 0; p < 2; ++p) {
            int v  = tid + p * 256;
            int mr = v & 31;
            int kk = (v >> 5) * 4;
            int gm = m0 + mr;
            v4h hv;
            hv[0] = (_Float16)fa[p].x; hv[1] = (_Float16)fa[p].y;
            hv[2] = (_Float16)fa[p].z; hv[3] = (_Float16)fa[p].w;
            if (gm >= M) { v4h z4 = {}; hv = z4; }
            *(v4h*)&Ash[buf][mr][kk] = hv;
        }
        #pragma unroll
        for (int p = 0; p < 8; ++p) {
            int v  = tid + p * 256;
            int n4 = (v & 31) * 4;
            int kk = v >> 5;
            Bsh[buf][n4 + 0][kk] = (_Float16)fb[p].x;
            Bsh[buf][n4 + 1][kk] = (_Float16)fb[p].y;
            Bsh[buf][n4 + 2][kk] = (_Float16)fb[p].z;
            Bsh[buf][n4 + 3][kk] = (_Float16)fb[p].w;
        }
    };
    // ---- 2 K-steps x 4 N-subtiles = 8 wmma on buffer `buf` ----
    auto compute = [&](int buf) {
        #pragma unroll
        for (int s = 0; s < 2; ++s) {
            int ks  = s * 32;
            int kk0 = ks + hs * 8;
            v8h a0 = *(const v8h*)&Ash[buf][arow][kk0];
            v8h a1 = *(const v8h*)&Ash[buf][arow][kk0 + 16];
            v16h af = __builtin_shufflevector(a0, a1,
                        0,1,2,3,4,5,6,7,8,9,10,11,12,13,14,15);
            int kb0 = ks + hs * 8;
            #pragma unroll
            for (int q = 0; q < 4; ++q) {
                int bcol = wn + q * 16 + lr;
                v8h b0 = *(const v8h*)&Bsh[buf][bcol][kb0];
                v8h b1 = *(const v8h*)&Bsh[buf][bcol][kb0 + 16];
                v16h bf = __builtin_shufflevector(b0, b1,
                            0,1,2,3,4,5,6,7,8,9,10,11,12,13,14,15);
                acc[q] = __builtin_amdgcn_wmma_f32_16x16x32_f16(
                             false, af, false, bf, (short)0, acc[q], false, false);
            }
        }
    };

    // ---- pipelined main loop ----
    int nk = Kd / GTK;
    loadAB(0);
    storeAB(0);
    __syncthreads();
    for (int ki = 0; ki < nk; ++ki) {
        int cur  = ki & 1;
        bool more = (ki + 1) < nk;
        if (more) loadAB((ki + 1) * GTK);   // global loads fly behind wmma
        compute(cur);
        if (more) storeAB(cur ^ 1);
        __syncthreads();
    }

    // ---- store: VGPR r, lane -> (M = row0+r, N = col) ----
    int row0 = m0 + wm + hs * 8;
    #pragma unroll
    for (int q = 0; q < 4; ++q) {
        int col = n0 + wn + q * 16 + lr;
        #pragma unroll
        for (int r = 0; r < 8; ++r) {
            int row = row0 + r;
            if (row < M) {
                float vv = acc[q][r];
                if (bias) vv += bias[row];
                if (act)  vv = dsilu(vv);
                Cb[(long)row * ldc + col] = vv;
            }
        }
    }
}

// ---------------------------------------------------------------------------
// dar: g[b,c] = mean |x2-x1|
// ---------------------------------------------------------------------------
__global__ __launch_bounds__(256)
void k_absmean(const float* __restrict__ x1, const float* __restrict__ x2,
               float* __restrict__ g, int HW)
{
    int bc = blockIdx.x;
    const float* a = x1 + (long)bc * HW;
    const float* b = x2 + (long)bc * HW;
    float s = 0.f;
    for (int i = threadIdx.x; i < HW; i += 256) s += fabsf(b[i] - a[i]);
    __shared__ float sh[256];
    sh[threadIdx.x] = s; __syncthreads();
    for (int o = 128; o > 0; o >>= 1) {
        if (threadIdx.x < o) sh[threadIdx.x] += sh[threadIdx.x + o];
        __syncthreads();
    }
    if (threadIdx.x == 0) g[bc] = sh[0] / HW;
}

__global__ __launch_bounds__(256)
void k_att(const float* __restrict__ g, const float* __restrict__ W,
           const float* __restrict__ b, float* __restrict__ att)
{
    int t = threadIdx.x;              // 0..255 -> (b,o)
    int bI = t >> 7, o = t & 127;
    const float* gr = g + bI * Cc;
    float a = b[o];
    for (int c = 0; c < Cc; ++c) a += gr[c] * W[o * Cc + c];
    att[t] = dsigmoid(a);
}

__global__ __launch_bounds__(256)
void k_xmix(const float* __restrict__ x1, const float* __restrict__ x2,
            const float* __restrict__ att, float* __restrict__ xch,
            float* __restrict__ ad, long total, int HW)
{
    long idx = (long)blockIdx.x * 256 + threadIdx.x;
    if (idx >= total) return;
    long bc = idx / HW;
    float a  = att[bc];
    float v1 = x1[idx], v2 = x2[idx];
    xch[idx] = v1 * a + v2 * (1.f - a);
    ad[idx]  = fabsf(v2 - v1);
}

// ---------------------------------------------------------------------------
// per-(batch,channel) spatial stats: mean and v = sum((x-mu)^2)/(n-1)
// ---------------------------------------------------------------------------
__global__ __launch_bounds__(256)
void k_chanstats(const float* __restrict__ x, float* __restrict__ mean,
                 float* __restrict__ v, int HW)
{
    int bc = blockIdx.x;
    const float* xb = x + (long)bc * HW;
    float s = 0.f, q = 0.f;
    for (int i = threadIdx.x; i < HW; i += 256) { float t = xb[i]; s += t; q += t * t; }
    __shared__ float ss[256], sq[256];
    ss[threadIdx.x] = s; sq[threadIdx.x] = q; __syncthreads();
    for (int o = 128; o > 0; o >>= 1) {
        if (threadIdx.x < o) {
            ss[threadIdx.x] += ss[threadIdx.x + o];
            sq[threadIdx.x] += sq[threadIdx.x + o];
        }
        __syncthreads();
    }
    if (threadIdx.x == 0) {
        float m = ss[0] / HW;
        mean[bc] = m;
        float sd = sq[0] - ss[0] * ss[0] / HW;
        v[bc] = sd / (HW - 1);
    }
}

__global__ __launch_bounds__(256)
void k_simam_inplace(float* __restrict__ x, const float* __restrict__ mean,
                     const float* __restrict__ v, int HW, long total)
{
    long idx = (long)blockIdx.x * 256 + threadIdx.x;
    if (idx >= total) return;
    long bc = idx / HW;
    float t = x[idx];
    float d = t - mean[bc];
    float e = d * d / (4.f * (v[bc] + SIMAM_LAM)) + 0.5f;
    x[idx] = t * dsigmoid(e);
}

// ---------------------------------------------------------------------------
// depthwise 3x3 on the im2cswin view (gathered from full image), SAME pad
// ---------------------------------------------------------------------------
__global__ __launch_bounds__(256)
void k_dwconv_win(const float* __restrict__ src, const float* __restrict__ w,
                  const float* __restrict__ bias, float* __restrict__ out,
                  int hsp, int wsp, int Hm, int Wm, int C, long total)
{
    long idx = (long)blockIdx.x * 256 + threadIdx.x;
    if (idx >= total) return;
    long t = idx;
    int ww = t % Wm; t /= Wm;
    int hh = t % Hm; t /= Hm;
    int c  = t % C;
    int bb = t / C;
    int bI = bb / (hsp * wsp);
    int i  = (bb / wsp) % hsp;
    int j  = bb % wsp;
    float acc = bias[c];
    #pragma unroll
    for (int dh = -1; dh <= 1; ++dh)
        #pragma unroll
        for (int dw = -1; dw <= 1; ++dw) {
            int h2 = hh + dh, w2 = ww + dw;
            if (h2 >= 0 && h2 < Hm && w2 >= 0 && w2 < Wm) {
                int gh = h2 * hsp + i, gw = w2 * wsp + j;
                acc += w[c * 9 + (dh + 1) * 3 + (dw + 1)]
                       * src[((long)bI * C + c) * HWc + gh * Wc + gw];
            }
        }
    out[idx] = acc;
}

// depthwise 3x3 on materialized tensor (first C channels, batch stride sbs), SiLU
__global__ __launch_bounds__(256)
void k_dw3_silu(const float* __restrict__ src, long sbs,
                const float* __restrict__ w, const float* __restrict__ bias,
                float* __restrict__ out, int C, int Hm, int Wm, long total)
{
    long idx = (long)blockIdx.x * 256 + threadIdx.x;
    if (idx >= total) return;
    long t = idx;
    int ww = t % Wm; t /= Wm;
    int hh = t % Hm; t /= Hm;
    int c  = t % C;
    int bn = t / C;
    const float* sb = src + (long)bn * sbs + (long)c * (Hm * Wm);
    float acc = bias[c];
    #pragma unroll
    for (int dh = -1; dh <= 1; ++dh)
        #pragma unroll
        for (int dw = -1; dw <= 1; ++dw) {
            int h2 = hh + dh, w2 = ww + dw;
            if (h2 >= 0 && h2 < Hm && w2 >= 0 && w2 < Wm)
                acc += w[c * 9 + (dh + 1) * 3 + (dw + 1)] * sb[h2 * Wm + w2];
        }
    out[idx] = dsilu(acc);
}

// xs[bn][k][d][l] for the 4 scan directions
__global__ __launch_bounds__(256)
void k_build_xs(const float* __restrict__ xp, float* __restrict__ xs,
                int Hm, int Wm, long total)
{
    int L = Hm * Wm;
    long idx = (long)blockIdx.x * 256 + threadIdx.x;
    if (idx >= total) return;
    long t = idx;
    int l = t % L; t /= L;
    int d = t % Dc;
    int bn = t / Dc;
    const float* xpb = xp + ((long)bn * Dc + d) * L;
    float v0 = xpb[l];
    float v1 = xpb[(l % Hm) * Wm + (l / Hm)];   // transposed stream
    long ob = (long)bn * Kc * Dc * L + (long)d * L;
    long ks = (long)Dc * L;
    xs[ob + 0 * ks + l]           = v0;
    xs[ob + 1 * ks + l]           = v1;
    xs[ob + 2 * ks + (L - 1 - l)] = v0;
    xs[ob + 3 * ks + (L - 1 - l)] = v1;
}

// dt = softplus(dts @ dt_w + dt_b)
__global__ __launch_bounds__(256)
void k_dt(const float* __restrict__ xdbl, const float* __restrict__ dt_w,
          const float* __restrict__ dt_b, float* __restrict__ dt,
          int L, long total)
{
    long idx = (long)blockIdx.x * 256 + threadIdx.x;
    if (idx >= total) return;
    long t = idx;
    int l = t % L; t /= L;
    int d = t % Dc; t /= Dc;
    int k = t % Kc;
    int bn = t / Kc;
    const float* dts = xdbl + ((long)(bn * Kc + k) * 36) * L;  // rows 0..3
    float a = dt_b[k * Dc + d];
    #pragma unroll
    for (int r = 0; r < Rc; ++r)
        a += dt_w[(k * Dc + d) * Rc + r] * dts[(long)r * L + l];
    dt[idx] = (a > 20.f) ? a : log1pf(__expf(a));
}

// selective scan: 256-thread block handles 4 (bn,k) groups; one thread per d
#define SCAN_CH 32
__global__ __launch_bounds__(256)
void k_scan(const float* __restrict__ xs, const float* __restrict__ dt,
            const float* __restrict__ xdbl, const float* __restrict__ A_log,
            const float* __restrict__ Ds, float* __restrict__ ys, int L)
{
    __shared__ float Bsh[4][Nst][SCAN_CH];
    __shared__ float Csh[4][Nst][SCAN_CH];
    int sub = threadIdx.x >> 6;          // 0..3
    int d   = threadIdx.x & 63;
    int blk = blockIdx.x * 4 + sub;      // bn*4 + k
    int k   = blk & 3;
    long base = (long)blk * Dc * L + (long)d * L;
    const float* xsd = xs + base;
    const float* dtd = dt + base;
    float*       ysd = ys + base;
    const float* Brow = xdbl + (long)blk * 36 * L + 4  * L;
    const float* Crow = xdbl + (long)blk * 36 * L + 20 * L;

    float Aa[Nst], h[Nst];
    #pragma unroll
    for (int n = 0; n < Nst; ++n) {
        Aa[n] = -__expf(A_log[(k * Dc + d) * Nst + n]);
        h[n] = 0.f;
    }
    float Dv = Ds[k * Dc + d];

    for (int l0 = 0; l0 < L; l0 += SCAN_CH) {
        __syncthreads();
        for (int i = d; i < Nst * SCAN_CH; i += 64) {
            int n = i >> 5, c = i & (SCAN_CH - 1);
            Bsh[sub][n][c] = Brow[(long)n * L + l0 + c];
            Csh[sub][n][c] = Crow[(long)n * L + l0 + c];
        }
        __syncthreads();
        if (l0 + SCAN_CH < L) {
            // speculative prefetch of next chunk's B/C rows -> global_prefetch_b8
            __builtin_prefetch(&Brow[(long)(d & 15) * L + l0 + SCAN_CH], 0, 1);
            __builtin_prefetch(&Crow[(long)(d & 15) * L + l0 + SCAN_CH], 0, 1);
        }
        for (int c = 0; c < SCAN_CH; ++c) {
            int l = l0 + c;
            float dtv = dtd[l];
            float xv  = xsd[l];
            float dx  = dtv * xv;
            float y = 0.f;
            #pragma unroll
            for (int n = 0; n < Nst; ++n) {
                h[n] = __expf(dtv * Aa[n]) * h[n] + dx * Bsh[sub][n][c];
                y += h[n] * Csh[sub][n][c];
            }
            ysd[l] = y + Dv * xv;
        }
    }
}

// combine 4 scan directions back into y[bn][d][l]
__global__ __launch_bounds__(256)
void k_combine(const float* __restrict__ ys, float* __restrict__ y,
               int Hm, int Wm, long total)
{
    int L = Hm * Wm;
    long idx = (long)blockIdx.x * 256 + threadIdx.x;
    if (idx >= total) return;
    long t = idx;
    int l = t % L; t /= L;
    int d = t % Dc;
    int bn = t / Dc;
    const float* yb = ys + (long)bn * Kc * Dc * L + (long)d * L;
    long ks = (long)Dc * L;
    int hh = l / Wm, ww = l % Wm;
    int lt = ww * Hm + hh;
    float v = yb[l] + yb[2 * ks + (L - 1 - l)]
            + yb[1 * ks + lt] + yb[3 * ks + (L - 1 - lt)];
    y[((long)bn * Dc + d) * L + l] = v;
}

// ---------------------------------------------------------------------------
// ln2d across channels per spatial position (+ optional x2 add,
// optional *silu(z), optional silu on output)
// ---------------------------------------------------------------------------
__global__ __launch_bounds__(256)
void k_ln(const float* __restrict__ x, long xbs, const float* __restrict__ x2,
          float* __restrict__ out, long obs,
          const float* __restrict__ g, const float* __restrict__ bt,
          const float* __restrict__ z, long zbs,
          int C, int HW, int act, long total)
{
    long idx = (long)blockIdx.x * 256 + threadIdx.x;
    if (idx >= total) return;
    long b = idx / HW;
    int l = idx % HW;
    const float* xb  = x + b * xbs + l;
    const float* x2b = x2 ? x2 + b * xbs + l : nullptr;
    float s = 0.f, q = 0.f;
    for (int c = 0; c < C; ++c) {
        float t = xb[(long)c * HW];
        if (x2b) t += x2b[(long)c * HW];
        s += t; q += t * t;
    }
    float m  = s / C;
    float rs = rsqrtf(q / C - m * m + LN_EPS);
    float*       ob = out + b * obs + l;
    const float* zb = z ? z + b * zbs + l : nullptr;
    for (int c = 0; c < C; ++c) {
        float t = xb[(long)c * HW];
        if (x2b) t += x2b[(long)c * HW];
        float v = (t - m) * rs * g[c] + bt[c];
        if (zb) v *= dsilu(zb[(long)c * HW]);
        if (act) v = dsilu(v);
        ob[(long)c * HW] = v;
    }
}

// t = branch * im2cswin(xcat)
__global__ __launch_bounds__(256)
void k_mul_view(const float* __restrict__ br, const float* __restrict__ xcat,
                float* __restrict__ out, int hsp, int wsp, int Hm, int Wm,
                long total)
{
    long idx = (long)blockIdx.x * 256 + threadIdx.x;
    if (idx >= total) return;
    long t = idx;
    int ww = t % Wm; t /= Wm;
    int hh = t % Hm; t /= Hm;
    int c  = t % Cc;
    int bb = t / Cc;
    int bI = bb / (hsp * wsp);
    int i  = (bb / wsp) % hsp;
    int j  = bb % wsp;
    float xv = xcat[((long)bI * Cc + c) * HWc + (hh * hsp + i) * Wc + (ww * wsp + j)];
    out[idx] = br[idx] * xv;
}

// simam(t) * x1branch, scattered back to full image (cswin2im), += if accum
__global__ __launch_bounds__(256)
void k_simam_scatter(const float* __restrict__ t, const float* __restrict__ mean,
                     const float* __restrict__ v, const float* __restrict__ x1br,
                     float* __restrict__ xsp, int hsp, int wsp, int Hm, int Wm,
                     int accum, long total)
{
    long idx = (long)blockIdx.x * 256 + threadIdx.x;
    if (idx >= total) return;
    long tt = idx;
    int ww = tt % Wm; tt /= Wm;
    int hh = tt % Hm; tt /= Hm;
    int c  = tt % Cc;
    int bb = tt / Cc;
    int bc = bb * Cc + c;
    float x = t[idx];
    float d = x - mean[bc];
    float e = d * d / (4.f * (v[bc] + SIMAM_LAM)) + 0.5f;
    float r = x * dsigmoid(e) * x1br[idx];
    int bI = bb / (hsp * wsp);
    int i  = (bb / wsp) % hsp;
    int j  = bb % wsp;
    long o = ((long)bI * Cc + c) * HWc + (hh * hsp + i) * Wc + (ww * wsp + j);
    if (accum) xsp[o] += r; else xsp[o] = r;
}

// oproj_w (M,64,128) -> (M,128,64)
__global__ __launch_bounds__(256)
void k_transpose_oproj(const float* __restrict__ w, float* __restrict__ wt)
{
    long idx = (long)blockIdx.x * 256 + threadIdx.x;
    if (idx >= 4L * Dc * Cc) return;
    long t = idx;
    int c = t % Cc; t /= Cc;
    int d = t % Dc;
    int m = t / Dc;
    wt[((long)m * Cc + c) * Dc + d] = w[idx];
}

// ---------------------------------------------------------------------------
static inline int nblk(long n) { return (int)((n + 255) / 256); }

extern "C" void kernel_launch(void* const* d_in, const int* in_sizes, int n_in,
                              void* d_out, int out_size, void* d_ws, size_t ws_size,
                              hipStream_t stream)
{
    const float* x1       = (const float*)d_in[0];
    const float* x2       = (const float*)d_in[1];
    const float* dw_w     = (const float*)d_in[2];
    const float* dw_b     = (const float*)d_in[3];
    const float* ssc_w    = (const float*)d_in[4];
    const float* ssc_b    = (const float*)d_in[5];
    const float* xproj_w  = (const float*)d_in[6];
    const float* dt_w     = (const float*)d_in[7];
    const float* dt_b     = (const float*)d_in[8];
    const float* A_log    = (const float*)d_in[9];
    const float* Ds       = (const float*)d_in[10];
    const float* onorm_g  = (const float*)d_in[11];
    const float* onorm_b  = (const float*)d_in[12];
    const float* oproj_w  = (const float*)d_in[13];
    const float* oproj_b  = (const float*)d_in[14];
    const float* post_g   = (const float*)d_in[15];
    const float* post_b   = (const float*)d_in[16];
    const float* dar_w    = (const float*)d_in[17];
    const float* dar_b    = (const float*)d_in[18];
    const float* ch_w     = (const float*)d_in[19];
    const float* ch_b     = (const float*)d_in[20];
    const float* ch_g     = (const float*)d_in[21];
    const float* ch_beta  = (const float*)d_in[22];
    const float* sp_w     = (const float*)d_in[23];
    const float* sp_b     = (const float*)d_in[24];
    const float* sp_g     = (const float*)d_in[25];
    const float* sp_beta  = (const float*)d_in[26];
    const float* mlp_g    = (const float*)d_in[27];
    const float* mlp_beta = (const float*)d_in[28];
    const float* mlp_w1   = (const float*)d_in[29];
    const float* mlp_b1   = (const float*)d_in[30];
    const float* mlp_w2   = (const float*)d_in[31];
    const float* mlp_b2   = (const float*)d_in[32];
    float* outp = (float*)d_out;

    // ---- workspace carve (floats) ----
    float* W = (float*)d_ws;
    size_t off = 0;
    auto carve = [&](size_t n) { float* p = W + off; off += n; return p; };
    float* gbuf  = carve(256);
    float* att   = carve(256);
    float* meanb = carve(2048);
    float* varb  = carve(2048);
    float* wtOp  = carve(4L * Cc * Dc);          // transposed oproj weights
    float* ad    = carve(2359296);               // |x2-x1| -> x_concat (in place)
    float* xch   = carve(2359296);               // x_channels
    float* brout = carve(4L * 2359296);          // 4 branch outputs
    float* P     = carve(21600000);              // aliased big scratch region

    // branch-scratch aliases inside P
    float* xbr  = P;
    float* xp   = P + 2359296;
    float* xs   = P + 3538944;
    float* xdbl = P + 8257536;
    float* dtb  = P + 10911744;
    float* ysb  = P + 15630336;
    float* ybuf = P + 20348928;
    // tail aliases (branch scratch is dead by then)
    float* xsp  = P;
    float* tbuf = P + 2359296;
    float* chb  = P + 4718592;
    float* spb  = P + 7077888;
    float* hb   = P + 9437184;
    float* h4   = P + 11796480;

    auto gemm = [&](const float* A, int lda, long sA, int amod,
                    const float* Bm, int ldb, long sB,
                    float* Cm, int ldc, long sC,
                    const float* bias, int act,
                    int M, int N, int Kd, int batches) {
        dim3 g(N / GTN, (M + GTM - 1) / GTM, batches);
        k_wmma_gemm<<<g, 256, 0, stream>>>(A, lda, sA, amod, Bm, ldb, sB,
                                           Cm, ldc, sC, bias, act, M, N, Kd);
    };

    // ---- dar + simam(|x2-x1|) ----
    k_absmean<<<Bc * Cc, 256, 0, stream>>>(x1, x2, gbuf, HWc);
    k_att<<<1, 256, 0, stream>>>(gbuf, dar_w, dar_b, att);
    k_xmix<<<nblk(2359296), 256, 0, stream>>>(x1, x2, att, xch, ad, 2359296, HWc);
    k_chanstats<<<Bc * Cc, 256, 0, stream>>>(ad, meanb, varb, HWc);
    k_simam_inplace<<<nblk(2359296), 256, 0, stream>>>(ad, meanb, varb, HWc, 2359296);
    k_transpose_oproj<<<nblk(4L * Dc * Cc), 256, 0, stream>>>(oproj_w, wtOp);

    // ---- 4 branches ----
    for (int m = 0; m < 4; ++m) {
        int hsp = (m & 1) ? 8 : 1;
        int wsp = (m & 1) ? 1 : 8;
        int Hm = Hc / hsp, Wm = Wc / wsp;
        const float* src = (m < 2) ? x1 : x2;

        long tot_full = (long)BNc * Cc * Lc;     // 2 359 296
        long tot_half = (long)BNc * Dc * Lc;     // 1 179 648
        long tot_dt   = (long)BNc * Kc * Dc * Lc;

        k_dwconv_win<<<nblk(tot_full), 256, 0, stream>>>(
            src, dw_w + (long)m * Cc * 9, dw_b + m * Cc, xbr,
            hsp, wsp, Hm, Wm, Cc, tot_full);
        k_dw3_silu<<<nblk(tot_half), 256, 0, stream>>>(
            xbr, (long)Cc * Lc, ssc_w + (long)m * Dc * 9, ssc_b + m * Dc,
            xp, Dc, Hm, Wm, tot_half);
        k_build_xs<<<nblk(tot_half), 256, 0, stream>>>(xp, xs, Hm, Wm, tot_half);

        // x_dbl = xproj_w[k] (36x64) @ xs[bn,k] (64x1152), 64 batches, WMMA
        gemm(xproj_w + (long)m * Kc * 36 * Dc, Dc, (long)36 * Dc, Kc,
             xs, Lc, (long)Dc * Lc,
             xdbl, Lc, (long)36 * Lc,
             nullptr, 0, 36, Lc, Dc, BNc * Kc);

        k_dt<<<nblk(tot_dt), 256, 0, stream>>>(
            xdbl, dt_w + (long)m * Kc * Dc * Rc, dt_b + (long)m * Kc * Dc,
            dtb, Lc, tot_dt);
        k_scan<<<BNc, 256, 0, stream>>>(
            xs, dtb, xdbl, A_log + (long)m * Kc * Dc * Nst,
            Ds + (long)m * Kc * Dc, ysb, Lc);
        k_combine<<<nblk(tot_half), 256, 0, stream>>>(ysb, ybuf, Hm, Wm, tot_half);

        // ln2d over D + *silu(z)   (z = channels 64..127 of xbr)
        k_ln<<<nblk((long)BNc * Lc), 256, 0, stream>>>(
            ybuf, (long)Dc * Lc, nullptr, ybuf, (long)Dc * Lc,
            onorm_g + m * Dc, onorm_b + m * Dc,
            xbr + (long)Dc * Lc, (long)Cc * Lc,
            Dc, Lc, 0, (long)BNc * Lc);

        // oproj: (128x64) @ y (64x1152), 16 batches, WMMA
        float* bo = brout + (long)m * BNc * Cc * Lc;
        gemm(wtOp + (long)m * Cc * Dc, Dc, 0, 1,
             ybuf, Lc, (long)Dc * Lc,
             bo, Lc, (long)Cc * Lc,
             oproj_b + m * Cc, 0, Cc, Lc, Dc, BNc);

        // post ln2d (in place)
        k_ln<<<nblk((long)BNc * Lc), 256, 0, stream>>>(
            bo, (long)Cc * Lc, nullptr, bo, (long)Cc * Lc,
            post_g + m * Cc, post_b + m * Cc, nullptr, 0,
            Cc, Lc, 0, (long)BNc * Lc);
    }

    // ---- spatial fusion: simam(xc*x2_br) * x1_br, merged back ----
    long tot_full = (long)BNc * Cc * Lc;
    // w-direction: view (1,8), x2_w = brout[2], x1_w = brout[0]
    k_mul_view<<<nblk(tot_full), 256, 0, stream>>>(
        brout + 2L * BNc * Cc * Lc, ad, tbuf, 1, 8, 96, 12, tot_full);
    k_chanstats<<<BNc * Cc, 256, 0, stream>>>(tbuf, meanb, varb, Lc);
    k_simam_scatter<<<nblk(tot_full), 256, 0, stream>>>(
        tbuf, meanb, varb, brout, xsp, 1, 8, 96, 12, 0, tot_full);
    // h-direction: view (8,1), x2_h = brout[3], x1_h = brout[1]
    k_mul_view<<<nblk(tot_full), 256, 0, stream>>>(
        brout + 3L * BNc * Cc * Lc, ad, tbuf, 8, 1, 12, 96, tot_full);
    k_chanstats<<<BNc * Cc, 256, 0, stream>>>(tbuf, meanb, varb, Lc);
    k_simam_scatter<<<nblk(tot_full), 256, 0, stream>>>(
        tbuf, meanb, varb, brout + 1L * BNc * Cc * Lc, xsp, 8, 1, 12, 96, 1, tot_full);

    // ---- channel / spatial 1x1 conv paths (WMMA) + LN + SiLU ----
    gemm(ch_w, Cc, 0, 1, xch, HWc, (long)Cc * HWc,
         chb, HWc, (long)Cc * HWc, ch_b, 0, Cc, HWc, Cc, Bc);
    k_ln<<<nblk((long)Bc * HWc), 256, 0, stream>>>(
        chb, (long)Cc * HWc, nullptr, chb, (long)Cc * HWc,
        ch_g, ch_beta, nullptr, 0, Cc, HWc, 1, (long)Bc * HWc);

    gemm(sp_w, Cc, 0, 1, xsp, HWc, (long)Cc * HWc,
         spb, HWc, (long)Cc * HWc, sp_b, 0, Cc, HWc, Cc, Bc);
    k_ln<<<nblk((long)Bc * HWc), 256, 0, stream>>>(
        spb, (long)Cc * HWc, nullptr, spb, (long)Cc * HWc,
        sp_g, sp_beta, nullptr, 0, Cc, HWc, 1, (long)Bc * HWc);

    // h = ln2d(ch + sp)
    k_ln<<<nblk((long)Bc * HWc), 256, 0, stream>>>(
        chb, (long)Cc * HWc, spb, hb, (long)Cc * HWc,
        mlp_g, mlp_beta, nullptr, 0, Cc, HWc, 0, (long)Bc * HWc);

    // MLP: 128 -> 512 (SiLU) -> 128 (writes d_out directly)
    gemm(mlp_w1, Cc, 0, 1, hb, HWc, (long)Cc * HWc,
         h4, HWc, (long)4 * Cc * HWc, mlp_b1, 1, 4 * Cc, HWc, Cc, Bc);
    gemm(mlp_w2, 4 * Cc, 0, 1, h4, HWc, (long)4 * Cc * HWc,
         outp, HWc, (long)Cc * HWc, mlp_b2, 0, Cc, HWc, 4 * Cc, Bc);
}
